// GRU2D_70763881168994
// MI455X (gfx1250) — matrix operand order repaired and checked by reading, not compiled, not run on hardware
//
#include <hip/hip_runtime.h>
#include <hip/hip_bf16.h>
#include <math.h>

// ---------------------------------------------------------------------------
// 2D snake-order GRU over 32x32 lattice, B=256, W=128.
// 1024 fully sequential cell steps; batch is the only parallelism.
// 16 workgroups x 16 batch samples; whole recurrence in one kernel.
// CDNA5-specific design:
//   * v_wmma_f32_16x16x32_f16 for both per-cell GEMMs (8 + 12 WMMA/cell)
//   * 320KB LDS/WGP: previous-row hidden state (128KB f16) fully LDS-resident
//   * GEMM2 tile mapping aligned with gate blocks (r|z|n): gate math + fc
//     partial dot products fully register-resident
//   * gate-affine constants folded & preloaded to registers (loop-invariant)
//   * 3 workgroup barriers per cell; zero global traffic on critical path
// ---------------------------------------------------------------------------

typedef __attribute__((ext_vector_type(16))) _Float16 v16h;
typedef __attribute__((ext_vector_type(8)))  _Float16 v8h;
typedef __attribute__((ext_vector_type(8)))  float    v8f;

#define NB   16     // batch samples per block
#define WD   128    // hidden width
#define LROW 32     // lattice size

// A-fragment (16x32 f16, row-major source, ld halfs per row), per ISA 7.12.2.
__device__ __forceinline__ v16h loadA(const _Float16* base, int ld, int row0,
                                      int kb, int lane) {
  int m  = row0 + (lane & 15);
  int kh = (lane >> 4) << 3;
  const _Float16* p = base + m * ld + kb + kh;
  v8h lo = *(const v8h*)(p);
  v8h hi = *(const v8h*)(p + 16);
  v16h r;
#pragma unroll
  for (int i = 0; i < 8; ++i) { r[i] = lo[i]; r[i + 8] = hi[i]; }
  return r;
}

// B-fragment (32x16 f16): source stored [n][K]; two 16B LDS reads (k = kb+8h.., kb+16+8h..).
__device__ __forceinline__ v16h loadB(const _Float16* rowptr, int kb, int lane) {
  const _Float16* p = rowptr + kb + ((lane >> 4) << 3);
  v8h lo = *(const v8h*)(p);
  v8h hi = *(const v8h*)(p + 16);
  v16h r;
#pragma unroll
  for (int i = 0; i < 8; ++i) { r[i] = lo[i]; r[i + 8] = hi[i]; }
  return r;
}

__device__ __forceinline__ float sigmoidf_(float x) {
  return 1.0f / (1.0f + expf(-x));
}

__global__ void gru2d_convert_weights(const float* __restrict__ Wh,
                                      const float* __restrict__ Whh,
                                      _Float16* __restrict__ WhH,
                                      _Float16* __restrict__ WhhH) {
  int idx = blockIdx.x * blockDim.x + threadIdx.x;
  if (idx < WD * 2 * WD) WhH[idx]  = (_Float16)Wh[idx];    // 128*256
  if (idx < 3 * WD * WD) WhhH[idx] = (_Float16)Whh[idx];   // 384*128
}

__global__ __launch_bounds__(256) void gru2d_main(
    const float* __restrict__ x,      // (256,32,32)
    const float* __restrict__ Wx,     // (1,2)
    const float* __restrict__ W_ih,   // (384,2)
    const float* __restrict__ b_ih,   // (384)
    const float* __restrict__ b_hh,   // (384)
    const float* __restrict__ Wf,     // (2,128)
    const float* __restrict__ bf,     // (2)
    const _Float16* __restrict__ WhH,   // (128,256) f16
    const _Float16* __restrict__ WhhH,  // (384,128) f16
    float* __restrict__ out)          // (256)
{
  // Large-LDS layout (CDNA5: up to 320KB per workgroup). ~146KB total.
  __shared__ _Float16 hprevL[LROW][NB][WD];  // 128KB: prev-row hidden (f16)
  __shared__ _Float16 hnbH[NB][WD];          // 4KB: in-row neighbor h (B op)
  __shared__ _Float16 hintH[NB][WD];         // 4KB: h_inter f16 (B op GEMM2)
  __shared__ float    hintF[WD][NB];         // 8KB: h_inter f32 (GRU h input)
  __shared__ float    fcpart[2][8][2][NB];   // 2KB: fc partial dots
  __shared__ float    pbuf[NB], xcurL[NB], xnb[NB];
  __shared__ float    accL[NB], aaL[NB], lplast[NB], mlast[NB];

  const int tid  = threadIdx.x;
  const int lane = tid & 31;
  const int wave = tid >> 5;
  const int n16  = lane & 15;
  const int hi2  = lane >> 4;         // half-selector (0/1)
  const int hi8  = hi2 << 3;
  const int b0   = blockIdx.x * NB;

  // one-time init: zero hprevL (row-0 up-neighbor is zero) and accumulators
  {
    _Float16* hp = &hprevL[0][0][0];
    for (int e = tid; e < LROW * NB * WD; e += 256) hp[e] = (_Float16)0.f;
    if (tid < NB) { accL[tid] = 0.f; aaL[tid] = 0.f; }
  }

  // loop-invariant gate-affine constants, register-resident per owned element:
  //   r/z:  arg = acc[v] + (W0-W1)*p + (W1 + b_ih + b_hh)
  //   n:    arg = (W0-W1)*p + (W1 + b_ih) + r * (acc[v] + b_hh)
  float cD0[8], cC0[8], cD1[8], cC1[8], cD2[8], cC2[8], cBN[8], wf0v[8], wf1v[8];
#pragma unroll
  for (int v = 0; v < 8; ++v) {
    int m = wave * 16 + hi8 + v;
    float w0 = W_ih[2 * m],            w1 = W_ih[2 * m + 1];
    cD0[v] = w0 - w1;  cC0[v] = w1 + b_ih[m] + b_hh[m];
    float z0 = W_ih[2 * (WD + m)],     z1 = W_ih[2 * (WD + m) + 1];
    cD1[v] = z0 - z1;  cC1[v] = z1 + b_ih[WD + m] + b_hh[WD + m];
    float n0 = W_ih[2 * (2 * WD + m)], n1 = W_ih[2 * (2 * WD + m) + 1];
    cD2[v] = n0 - n1;  cC2[v] = n1 + b_ih[2 * WD + m];
    cBN[v] = b_hh[2 * WD + m];
    wf0v[v] = Wf[m];  wf1v[v] = Wf[WD + m];
  }
  const float Wx0 = Wx[0], Wx1 = Wx[1];
  const float bf0 = bf[0], bf1 = bf[1];
  __syncthreads();

  for (int i = 0; i < LROW; ++i) {
    for (int t = 0; t < LROW; ++t) {
      const int j = (i & 1) ? (LROW - 1 - t) : t;   // snake order

      // ---- emb scalar p + stage current spin (overlaps GEMM1) ----
      if (tid < NB) {
        int n = tid;
        float xn  = (t == 0) ? 0.f : xnb[n];                  // row-start reset
        float xup = (i > 0) ? x[(size_t)(b0 + n) * 1024 + (i - 1) * 32 + j] : 0.f;
        float xc  = x[(size_t)(b0 + n) * 1024 + i * 32 + j];
        xcurL[n]  = xc;
        float xi  = Wx0 * xn + Wx1 * xup;
        pbuf[n]   = (xi + 1.0f) * 0.5f;
      }

      // ---- GEMM1: h_inter = Wh @ [h_nb ; h_up] (one 16x16 tile per wave).
      //      h_nb == 0 at row start -> skip lower-half WMMAs (uniform branch).
      {
        v8f acc = {};
        const _Float16* bUp = &hprevL[j][n16][0];
        v16h bu[4];
#pragma unroll
        for (int kk = 0; kk < 4; ++kk) bu[kk] = loadB(bUp, kk * 32, lane);
        if (t > 0) {
          const _Float16* bLow = &hnbH[n16][0];
          v16h bl[4];
#pragma unroll
          for (int kk = 0; kk < 4; ++kk) bl[kk] = loadB(bLow, kk * 32, lane);
#pragma unroll
          for (int kk = 0; kk < 4; ++kk) {
            v16h a = loadA(WhH, 2 * WD, wave * 16, kk * 32, lane);
            acc = __builtin_amdgcn_wmma_f32_16x16x32_f16(
                false, a, false, bl[kk], (short)0, acc, false, false);
          }
        }
#pragma unroll
        for (int kk = 0; kk < 4; ++kk) {
          v16h a = loadA(WhH, 2 * WD, wave * 16, WD + kk * 32, lane);
          acc = __builtin_amdgcn_wmma_f32_16x16x32_f16(
              false, a, false, bu[kk], (short)0, acc, false, false);
        }
        int mbase = wave * 16 + hi8;
#pragma unroll
        for (int v = 0; v < 8; ++v) {
          int m = mbase + v;
          hintF[m][n16] = acc[v];
          hintH[n16][m] = (_Float16)acc[v];
        }
      }
      __syncthreads();   // barrier 1: hintH/hintF visible

      // ---- GEMM2: gh(r|z|n) rows 16w..16w+15; B fragment reused x3 ----
      v8f aR = {}, aZ = {}, aN = {};
      {
        const _Float16* bI = &hintH[n16][0];
        v16h bi[4];
#pragma unroll
        for (int kk = 0; kk < 4; ++kk) bi[kk] = loadB(bI, kk * 32, lane);
#pragma unroll
        for (int kk = 0; kk < 4; ++kk) {
          int kb = kk * 32;
          v16h ar = loadA(WhhH, WD, wave * 16,          kb, lane);
          v16h az = loadA(WhhH, WD, WD + wave * 16,     kb, lane);
          v16h an = loadA(WhhH, WD, 2 * WD + wave * 16, kb, lane);
          aR = __builtin_amdgcn_wmma_f32_16x16x32_f16(
              false, ar, false, bi[kk], (short)0, aR, false, false);
          aZ = __builtin_amdgcn_wmma_f32_16x16x32_f16(
              false, az, false, bi[kk], (short)0, aZ, false, false);
          aN = __builtin_amdgcn_wmma_f32_16x16x32_f16(
              false, an, false, bi[kk], (short)0, aN, false, false);
        }
      }

      // ---- GRU gates + fc partials, register-resident ----
      {
        float p = pbuf[n16];
        float f0 = 0.f, f1 = 0.f;
#pragma unroll
        for (int v = 0; v < 8; ++v) {
          int m = wave * 16 + hi8 + v;
          float r  = sigmoidf_(aR[v] + cD0[v] * p + cC0[v]);
          float z  = sigmoidf_(aZ[v] + cD1[v] * p + cC1[v]);
          float nn = tanhf(cD2[v] * p + cC2[v] + r * (aN[v] + cBN[v]));
          float h  = hintF[m][n16];
          float hnew = (1.0f - z) * nn + z * h;
          _Float16 hh = (_Float16)hnew;
          hnbH[n16][m] = hh;            // in-row neighbor for next cell
          hprevL[j][n16][m] = hh;       // up-neighbor for row i+1 (same col)
          f0 += wf0v[v] * hnew;
          f1 += wf1v[v] * hnew;
        }
        fcpart[0][wave][hi2][n16] = f0;
        fcpart[1][wave][hi2][n16] = f1;
      }
      __syncthreads();   // barrier 2: hnbH/hprevL/fcpart visible

      // ---- logits reduce + log_softmax + lp accumulation ----
      if (tid < NB) {
        int n = tid;
        float l0 = bf0, l1 = bf1;
#pragma unroll
        for (int w = 0; w < 8; ++w) {
          l0 += fcpart[0][w][0][n] + fcpart[0][w][1][n];
          l1 += fcpart[1][w][0][n] + fcpart[1][w][1][n];
        }
        float mx  = fmaxf(l0, l1);
        float lse = mx + logf(expf(l0 - mx) + expf(l1 - mx));
        float y0 = l0 - lse, y1 = l1 - lse;
        float xc = xcurL[n];
        float mc = (1.0f + xc) * 0.5f;
        float lp = y0 * mc + y1 * (1.0f - mc);
        accL[n] += lp;
        aaL[n]  += mc;
        lplast[n] = lp;   // last processed cell is (31,0)
        mlast[n]  = mc;
        xnb[n]    = xc;   // in-row spin carry
      }
      __syncthreads();   // barrier 3: xnb/acc carried to next cell
    }
  }

  // aa==1 correction: replace lp(31,0) by lp*(1-m)  <=>  acc -= lp*m
  if (tid < NB) {
    int n = tid;
    float acc = accL[n];
    if (aaL[n] == 1.0f) acc -= lplast[n] * mlast[n];
    out[b0 + n] = acc;
  }
}

extern "C" void kernel_launch(void* const* d_in, const int* in_sizes, int n_in,
                              void* d_out, int out_size, void* d_ws, size_t ws_size,
                              hipStream_t stream) {
  (void)in_sizes; (void)n_in; (void)out_size; (void)ws_size;
  const float* x    = (const float*)d_in[0];
  const float* Wx   = (const float*)d_in[1];
  const float* Wh   = (const float*)d_in[2];
  const float* W_ih = (const float*)d_in[3];
  const float* Whh  = (const float*)d_in[4];
  const float* b_ih = (const float*)d_in[5];
  const float* b_hh = (const float*)d_in[6];
  const float* Wf   = (const float*)d_in[7];
  const float* bf   = (const float*)d_in[8];
  float* out = (float*)d_out;

  // d_ws layout: Wh f16 (64KB) | Whh f16 (96KB)
  char* ws = (char*)d_ws;
  _Float16* WhH  = (_Float16*)(ws);
  _Float16* WhhH = (_Float16*)(ws + 65536);

  gru2d_convert_weights<<<192, 256, 0, stream>>>(Wh, Whh, WhH, WhhH);
  gru2d_main<<<16, 256, 0, stream>>>(x, Wx, W_ih, b_ih, b_hh, Wf, bf,
                                     WhH, WhhH, out);
}
